// Attention_23562190586278
// MI455X (gfx1250) — compile-verified, matrix operand order checked by reading, not run on hardware
//
#include <hip/hip_runtime.h>
#include <math.h>

typedef __attribute__((ext_vector_type(16))) _Float16 v16h;
typedef __attribute__((ext_vector_type(8)))  _Float16 v8h;
typedef __attribute__((ext_vector_type(4)))  _Float16 v4h;
typedef __attribute__((ext_vector_type(8)))  float    v8f;
typedef __attribute__((ext_vector_type(4)))  float    v4f;

#define SEQ  2048
#define DH   64
#define KT   64      // keys per mainloop iteration (one LDS tile)
#define KSTR 72      // padded halves per K-tile row  (64 d + 8 pad)
#define VSTR 72      // padded halves per V^T-tile row (64 k + 8 pad)

static __device__ __forceinline__ v8f wmma_f16(v16h a, v16h b, v8f c) {
  return __builtin_amdgcn_wmma_f32_16x16x32_f16(
      false, a, false, b, (short)0, c, false, false);
}

static __device__ __forceinline__ v16h cat16(v8h a, v8h b) {
  return __builtin_shufflevector(a, b, 0, 1, 2, 3, 4, 5, 6, 7,
                                 8, 9, 10, 11, 12, 13, 14, 15);
}

// Global f32 row -> 16 halves in WMMA operand layout (halves 0..7 = p[0..7],
// halves 8..15 = p[16..23]); used only for the per-wave Q^T load.
static __device__ __forceinline__ v16h load_rowseg(const float* __restrict__ p,
                                                   float scale) {
  const v4f* p4 = (const v4f*)p;
  v4f x0 = p4[0], x1 = p4[1], y0 = p4[4], y1 = p4[5];
  v16h r;
#pragma unroll
  for (int i = 0; i < 4; ++i) {
    r[i]      = (_Float16)(x0[i] * scale);
    r[4 + i]  = (_Float16)(x1[i] * scale);
    r[8 + i]  = (_Float16)(y0[i] * scale);
    r[12 + i] = (_Float16)(y1[i] * scale);
  }
  return r;
}

__global__ void __launch_bounds__(256)
fa_fwd_wmma(const float* __restrict__ Q, const float* __restrict__ K,
            const float* __restrict__ V, float* __restrict__ O) {
  // double-buffered f16 tiles: K row-major [k][d], V transposed [d][k]
  __shared__ _Float16 ldsK[2 * KT * KSTR];
  __shared__ _Float16 ldsV[2 * DH * VSTR];

  const int tid  = threadIdx.x;
  const int lane = tid & 31;
  const int wave = tid >> 5;
  const int l15  = lane & 15;   // operand column/row index
  const int hi   = lane >> 4;   // half-wave: selects K-column block

  const size_t base = (size_t)blockIdx.y * SEQ * DH;
  const int q0 = (blockIdx.x * 8 + wave) * 16;

  const float* qp = Q + base;
  const float* kp = K + base;
  const float* vp = V + base;

  // cooperative staging coordinates: rows tk+16*r, float4 column td4
  const int tk  = tid >> 4;        // 0..15
  const int td4 = (tid & 15) * 4;  // 0..60

  // ---- Q^T in WMMA-B layout, held all pass, pre-scaled by 1/sqrt(D) ----
  v16h qB[2];
#pragma unroll
  for (int c = 0; c < 2; ++c)
    qB[c] = load_rowseg(qp + (size_t)(q0 + l15) * DH + c * 32 + hi * 8, 0.125f);

  v8f o[4];
#pragma unroll
  for (int j = 0; j < 4; ++j)
#pragma unroll
    for (int r = 0; r < 8; ++r) o[j][r] = 0.0f;

  float m = -INFINITY, lsum = 0.0f;

  v4f kst[4], vst[4];  // per-thread global staging registers

#define LOAD_TILE(k0n)                                                      \
  do {                                                                      \
    _Pragma("unroll") for (int r = 0; r < 4; ++r) {                         \
      kst[r] = *(const v4f*)(kp + (size_t)((k0n) + tk + 16 * r) * DH + td4);\
      vst[r] = *(const v4f*)(vp + (size_t)((k0n) + tk + 16 * r) * DH + td4);\
    }                                                                       \
  } while (0)

#define STORE_TILE(s)                                                       \
  do {                                                                      \
    _Float16* Kl = ldsK + (s) * (KT * KSTR);                                \
    _Float16* Vl = ldsV + (s) * (DH * VSTR);                                \
    _Pragma("unroll") for (int r = 0; r < 4; ++r) {                         \
      v4h h;                                                                \
      _Pragma("unroll") for (int i = 0; i < 4; ++i)                         \
          h[i] = (_Float16)kst[r][i];                                       \
      *(v4h*)&Kl[(tk + 16 * r) * KSTR + td4] = h;                           \
      _Pragma("unroll") for (int i = 0; i < 4; ++i)  /* transpose V */      \
          Vl[(td4 + i) * VSTR + tk + 16 * r] = (_Float16)vst[r][i];         \
    }                                                                       \
  } while (0)

  LOAD_TILE(0);
  STORE_TILE(0);

  for (int k0 = 0; k0 < SEQ; k0 += KT) {
    const int s = (k0 >> 6) & 1;
    const bool more = (k0 + KT) < SEQ;
    if (more) LOAD_TILE(k0 + KT);  // overlap HBM/L2 latency with compute

    __syncthreads();  // tile s ready; prior reads of tile 1-s retired

    const _Float16* Kl = ldsK + s * (KT * KSTR);
    const _Float16* Vl = ldsV + s * (DH * VSTR);

    // ---- S^T = K * Q^T : four 16-key subtiles, K-dim = D = 2x32 ----
    v8f sc[4];
#pragma unroll
    for (int t = 0; t < 4; ++t) {
#pragma unroll
      for (int r = 0; r < 8; ++r) sc[t][r] = 0.0f;
#pragma unroll
      for (int c = 0; c < 2; ++c) {
        const _Float16* kr = Kl + (t * 16 + l15) * KSTR + c * 32 + hi * 8;
        v16h a = cat16(*(const v8h*)kr, *(const v8h*)(kr + 16));
        sc[t] = wmma_f16(a, qB[c], sc[t]);
      }
    }

    // ---- online softmax (lane owns query column l15) ----
    float tmax = -INFINITY;
#pragma unroll
    for (int t = 0; t < 4; ++t)
#pragma unroll
      for (int r = 0; r < 8; ++r) tmax = fmaxf(tmax, sc[t][r]);
    tmax = fmaxf(tmax, __shfl_xor(tmax, 16, 32));

    const float mnew  = fmaxf(m, tmax);
    const float alpha = __expf(m - mnew);
    m = mnew;

    // exp() lands directly in WMMA-B layout: S^T C-layout VGPR r == operand
    // half r for the same key row. pB[0]=keys 0..31, pB[1]=keys 32..63.
    v16h pB[2];
    float rsum = 0.0f;
#pragma unroll
    for (int r = 0; r < 8; ++r) {
      float e0 = __expf(sc[0][r] - mnew);
      float e1 = __expf(sc[1][r] - mnew);
      float e2 = __expf(sc[2][r] - mnew);
      float e3 = __expf(sc[3][r] - mnew);
      rsum += (e0 + e1) + (e2 + e3);
      pB[0][r] = (_Float16)e0;  pB[0][8 + r] = (_Float16)e1;
      pB[1][r] = (_Float16)e2;  pB[1][8 + r] = (_Float16)e3;
    }
    rsum += __shfl_xor(rsum, 16, 32);
    lsum = lsum * alpha + rsum;

    // ---- O^T = alpha*O^T + V^T * P^T ----
#pragma unroll
    for (int j = 0; j < 4; ++j) {
#pragma unroll
      for (int r = 0; r < 8; ++r) o[j][r] *= alpha;
      const _Float16* vr = Vl + (j * 16 + l15) * VSTR;
      v16h a0 = cat16(*(const v8h*)(vr + hi * 8),
                      *(const v8h*)(vr + 16 + hi * 8));
      v16h a1 = cat16(*(const v8h*)(vr + 32 + hi * 8),
                      *(const v8h*)(vr + 48 + hi * 8));
      o[j] = wmma_f16(a0, pB[0], o[j]);
      o[j] = wmma_f16(a1, pB[1], o[j]);
    }

    if (more) STORE_TILE(1 - s);  // fill the other buffer for next iter
  }

  // ---- normalize and store ----
  const float inv = 1.0f / lsum;
  float* orow = O + base + (size_t)(q0 + l15) * DH;
#pragma unroll
  for (int j = 0; j < 4; ++j)
#pragma unroll
    for (int r = 0; r < 8; ++r)
      orow[j * 16 + hi * 8 + r] = o[j][r] * inv;

#undef LOAD_TILE
#undef STORE_TILE
}

extern "C" void kernel_launch(void* const* d_in, const int* in_sizes, int n_in,
                              void* d_out, int out_size, void* d_ws, size_t ws_size,
                              hipStream_t stream) {
  const float* q = (const float*)d_in[0];
  const float* k = (const float*)d_in[1];
  const float* v = (const float*)d_in[2];
  float* out = (float*)d_out;

  const int bh = in_sizes[0] / (SEQ * DH);  // B*H = 64
  dim3 grid(SEQ / (16 * 8), bh);            // (16, 64)
  dim3 block(256);                          // 8 waves, one 16-query tile each
  fa_fwd_wmma<<<grid, block, 0, stream>>>(q, k, v, out);
}
